// KineticsRNN_22127671509716
// MI455X (gfx1250) — matrix operand order repaired and checked by reading, not compiled, not run on hardware
//
#include <hip/hip_runtime.h>
#include <hip/hip_bf16.h>
#include <math.h>

typedef __attribute__((ext_vector_type(16))) _Float16 v16h;
typedef __attribute__((ext_vector_type(8)))  float    v8f;

#define BATCH   512
#define KSTEPS  512
#define HDIM    128
#define NTILES  24      // 384/16

static __device__ __forceinline__ unsigned pack2h(float a, float b) {
  _Float16 ha = (_Float16)a, hb = (_Float16)b;
  unsigned short ua = __builtin_bit_cast(unsigned short, ha);
  unsigned short ub = __builtin_bit_cast(unsigned short, hb);
  return (unsigned)ua | ((unsigned)ub << 16);
}
static __device__ __forceinline__ void unpack2h(unsigned d, float& a, float& b) {
  a = (float)__builtin_bit_cast(_Float16, (unsigned short)(d & 0xffffu));
  b = (float)__builtin_bit_cast(_Float16, (unsigned short)(d >> 16));
}
static __device__ __forceinline__ float sigmoidf_(float x) { return 1.0f / (1.0f + __expf(-x)); }
static __device__ __forceinline__ float tanh_fast(float x) { return 1.0f - 2.0f / (__expf(2.0f * x) + 1.0f); }

static __device__ __forceinline__ void rhs5(const float* y, const float* th, float* d) {
  float r1 = th[0] * y[0] * y[1];
  float r2 = th[1] * y[2];
  float r3 = th[2] * y[2] * y[3];
  float r4 = th[3] * y[4];
  float r5 = th[4] * y[0];
  d[0] = -r1 - r5;
  d[1] = -r1 + r2;
  d[2] =  r1 - r2 - r3;
  d[3] = -r3 + r4;
  d[4] =  r3 - r4 + r5;
}

__global__ __launch_bounds__(256) void kinetics_rnn_kernel(
    const float* __restrict__ y0,      // (512,3)
    const float* __restrict__ u_seq,   // (512,512,3)
    const float* __restrict__ dt_seq,  // (512,512)
    const float* __restrict__ y_seq,   // (512,512,3)
    const float* __restrict__ W_lift,  // (32,6)
    const float* __restrict__ b_lift,  // (32)
    const float* __restrict__ W_ih,    // (384,32)
    const float* __restrict__ W_hh,    // (384,128)
    const float* __restrict__ b_ih,    // (384)
    const float* __restrict__ b_hh,    // (384)
    const float* __restrict__ W_head,  // (5,128)
    const float* __restrict__ b_head,  // (5)
    const float* __restrict__ u2y,     // (3,3)
    float* __restrict__ out_yhat,      // (512,512,3)
    float* __restrict__ out_theta)     // (512,512,5)
{
  // ---- LDS (≈151 KB total) ----
  __shared__ alignas(32) unsigned whhP[NTILES * 4 * 32 * 8]; // 96 KB  f16 W_hh, WMMA B layout
  __shared__ alignas(32) unsigned wihP[NTILES * 32 * 8];     // 24 KB  f16 W_ih, WMMA B layout
  __shared__ alignas(32) unsigned wheadP[4 * 32 * 8];        //  4 KB  f16 W_head (N padded to 16), B layout
  __shared__ alignas(32) unsigned hA[4 * 32 * 8];            //  4 KB  f16 h, WMMA A layout (4 k-tiles)
  __shared__ alignas(32) unsigned liftA[32 * 8];             //  1 KB  f16 lift, WMMA A layout
  __shared__ alignas(16) _Float16 rzT[256 * 16];             //  8 KB  post-sigmoid r|z, [col][m]
  __shared__ alignas(16) _Float16 xnT[128 * 16];             //  4 KB  gx n-gate preact, [col][m]
  __shared__ alignas(16) _Float16 hnT[128 * 16];             //  4 KB  gh n-gate preact, [col][m]
  __shared__ float bSumS[256];                               // b_ih+b_hh for r/z cols
  __shared__ float bihN[128], bhhN[128];                     // n-gate biases
  __shared__ float bheadS[8];
  __shared__ float u2yS[9];
  __shared__ float yfull[16 * 5];
  __shared__ float thetaS[16 * 5];

  const int t    = threadIdx.x;
  const int lane = t & 31;
  const int wave = __builtin_amdgcn_readfirstlane(t >> 5);  // SGPR: scalar branches around WMMA
  const int bg   = blockIdx.x;          // batch tile: rows bg*16 .. bg*16+15

  // ---- one-time: pack weights into LDS in WMMA lane layouts ----
  // B operand (f16, 16x16x32): lane L, dword v -> N = L%16, K = (L<16?0:16)+2v, +1
  for (int idx = t; idx < NTILES * 32 * 8; idx += 256) {
    int v = idx & 7, ln = (idx >> 3) & 31, n = idx >> 8;
    int ncol = n * 16 + (ln & 15);
    int kk = ((ln < 16) ? 0 : 16) + 2 * v;
    wihP[idx] = pack2h(W_ih[ncol * 32 + kk], W_ih[ncol * 32 + kk + 1]);
  }
  for (int idx = t; idx < NTILES * 4 * 32 * 8; idx += 256) {
    int v = idx & 7, ln = (idx >> 3) & 31, kt = (idx >> 8) & 3, n = idx >> 10;
    int ncol = n * 16 + (ln & 15);
    int kk = kt * 32 + ((ln < 16) ? 0 : 16) + 2 * v;
    whhP[idx] = pack2h(W_hh[ncol * HDIM + kk], W_hh[ncol * HDIM + kk + 1]);
  }
  for (int idx = t; idx < 4 * 32 * 8; idx += 256) {          // W_head, N-padded 5->16
    int v = idx & 7, ln = (idx >> 3) & 31, kt = idx >> 8;
    int o = ln & 15;
    int kk = kt * 32 + ((ln < 16) ? 0 : 16) + 2 * v;
    wheadP[idx] = (o < 5) ? pack2h(W_head[o * HDIM + kk], W_head[o * HDIM + kk + 1]) : 0u;
  }
  for (int i = t; i < 256; i += 256) bSumS[i] = b_ih[i] + b_hh[i];
  if (t < 128) { bihN[t] = b_ih[256 + t]; bhhN[t] = b_hh[256 + t]; }
  if (t < 5) bheadS[t] = b_head[t];
  if (t < 9) u2yS[t] = u2y[t];
  for (int i = t; i < 4 * 32 * 8; i += 256) hA[i] = 0u;      // h0 = 0
  if (t < 16) {
    int b = bg * 16 + t;
    yfull[t * 5 + 0] = y0[b * 3 + 0];
    yfull[t * 5 + 1] = 0.f;
    yfull[t * 5 + 2] = y0[b * 3 + 1];
    yfull[t * 5 + 3] = 0.f;
    yfull[t * 5 + 4] = y0[b * 3 + 2];
  }

  // ---- per-thread loop invariants (live in VGPRs for all 512 steps) ----
  // Phase B: this thread produces liftA dword (lnB*8 + vB) = lift rows kkB0, kkB0+1 of batch row mB
  const int lnB = t >> 3, vB = t & 7;
  const int mB  = lnB & 15, hiB = lnB >> 4;
  const int kkB0 = (hiB << 4) | (vB << 1);
  float wl0[6], wl1[6];
  #pragma unroll
  for (int c = 0; c < 6; ++c) { wl0[c] = W_lift[kkB0 * 6 + c]; wl1[c] = W_lift[(kkB0 + 1) * 6 + c]; }
  const float bl0 = b_lift[kkB0], bl1 = b_lift[kkB0 + 1];
  const int liftIdx = lnB * 8 + vB;
  const float* uPtrB = u_seq + ((size_t)(bg * 16 + mB) * KSTEPS) * 3;

  // Phase D: hA dword index + transposed-column base per sub-iteration
  int dwD[4], jbD[4];
  #pragma unroll
  for (int i = 0; i < 4; ++i) {
    int p = t + 256 * i;
    int m = p & 15;
    int j = (p >> 4) << 1;
    int kt = j >> 5, kk = j & 31;
    int hi = kk >> 4;
    int v  = (kk & 15) >> 1;
    dwD[i] = (kt * 32 + hi * 16 + m) * 8 + v;
    jbD[i] = j * 16 + m;
  }

  // Phase F pointers (threads 0..15)
  const float* uPtrF  = u_seq  + ((size_t)(bg * 16 + (t & 15)) * KSTEPS) * 3;
  const float* dtPtrF = dt_seq + (size_t)(bg * 16 + (t & 15)) * KSTEPS;
  float*       yoPtrF = out_yhat + ((size_t)(bg * 16 + (t & 15)) * KSTEPS) * 3;

  const int mb = (lane >> 4) << 3;      // C layout: m = j + (lane>=16 ? 8 : 0)

  __syncthreads();

  int mod50 = 0;                        // k % 50, maintained incrementally
  for (int k = 0; k < KSTEPS; ++k) {
    const bool flag = (k > 0) && (mod50 == 0);   // uniform -> scalar branch

    // ---- Phase B: feat gather + lift = silu(feat @ W_lift^T + b) -> WMMA A layout ----
    {
      float f0 = uPtrB[0], f1 = uPtrB[1], f2 = uPtrB[2];
      float f3, f4, f5;
      if (flag) {
        const float* yp = &y_seq[((size_t)(bg * 16 + mB) * KSTEPS + (k - 1)) * 3];
        f3 = yp[0]; f4 = yp[1]; f5 = yp[2];
      } else {
        f3 = yfull[mB * 5 + 0]; f4 = yfull[mB * 5 + 2]; f5 = yfull[mB * 5 + 4];
      }
      float a0 = bl0 + f0*wl0[0] + f1*wl0[1] + f2*wl0[2] + f3*wl0[3] + f4*wl0[4] + f5*wl0[5];
      float a1 = bl1 + f0*wl1[0] + f1*wl1[1] + f2*wl1[2] + f3*wl1[3] + f4*wl1[4] + f5*wl1[5];
      liftA[liftIdx] = pack2h(a0 * sigmoidf_(a0), a1 * sigmoidf_(a1));
      uPtrB += 3;
    }
    __syncthreads();

    // ---- Phase C: WMMA gate GEMMs (scalar-uniform branching; EXEC all ones) ----
    {
      v16h aL  = *reinterpret_cast<const v16h*>(&liftA[lane * 8]);
      v16h aH0 = *reinterpret_cast<const v16h*>(&hA[(0 * 32 + lane) * 8]);
      v16h aH1 = *reinterpret_cast<const v16h*>(&hA[(1 * 32 + lane) * 8]);
      v16h aH2 = *reinterpret_cast<const v16h*>(&hA[(2 * 32 + lane) * 8]);
      v16h aH3 = *reinterpret_cast<const v16h*>(&hA[(3 * 32 + lane) * 8]);
      if (wave < 4) {
        // waves 0-3: 4 r/z tiles each (n = 0..15): one fused accumulator, sigmoid in-lane
        #pragma unroll
        for (int i = 0; i < 4; ++i) {
          int n = wave * 4 + i;
          int colG = n * 16 + (lane & 15);                 // 0..255
          v16h bI  = *reinterpret_cast<const v16h*>(&wihP[(n * 32 + lane) * 8]);
          v16h bH0 = *reinterpret_cast<const v16h*>(&whhP[((n * 4 + 0) * 32 + lane) * 8]);
          v16h bH1 = *reinterpret_cast<const v16h*>(&whhP[((n * 4 + 1) * 32 + lane) * 8]);
          v16h bH2 = *reinterpret_cast<const v16h*>(&whhP[((n * 4 + 2) * 32 + lane) * 8]);
          v16h bH3 = *reinterpret_cast<const v16h*>(&whhP[((n * 4 + 3) * 32 + lane) * 8]);
          float bias = bSumS[colG];
          v8f c; for (int j = 0; j < 8; ++j) c[j] = bias;
          c = __builtin_amdgcn_wmma_f32_16x16x32_f16(false, aL,  false, bI,  (short)0, c, false, false);
          c = __builtin_amdgcn_wmma_f32_16x16x32_f16(false, aH0, false, bH0, (short)0, c, false, false);
          c = __builtin_amdgcn_wmma_f32_16x16x32_f16(false, aH1, false, bH1, (short)0, c, false, false);
          c = __builtin_amdgcn_wmma_f32_16x16x32_f16(false, aH2, false, bH2, (short)0, c, false, false);
          c = __builtin_amdgcn_wmma_f32_16x16x32_f16(false, aH3, false, bH3, (short)0, c, false, false);
          unsigned* dst = reinterpret_cast<unsigned*>(&rzT[colG * 16 + mb]);
          dst[0] = pack2h(sigmoidf_(c[0]), sigmoidf_(c[1]));
          dst[1] = pack2h(sigmoidf_(c[2]), sigmoidf_(c[3]));
          dst[2] = pack2h(sigmoidf_(c[4]), sigmoidf_(c[5]));
          dst[3] = pack2h(sigmoidf_(c[6]), sigmoidf_(c[7]));
        }
      } else {
        // waves 4-7: 2 n-gate tiles each (n = 16..23): gx and gh kept separate
        #pragma unroll
        for (int i = 0; i < 2; ++i) {
          int n = 16 + (wave - 4) * 2 + i;
          int cn = (n - 16) * 16 + (lane & 15);            // 0..127
          v16h bI  = *reinterpret_cast<const v16h*>(&wihP[(n * 32 + lane) * 8]);
          v16h bH0 = *reinterpret_cast<const v16h*>(&whhP[((n * 4 + 0) * 32 + lane) * 8]);
          v16h bH1 = *reinterpret_cast<const v16h*>(&whhP[((n * 4 + 1) * 32 + lane) * 8]);
          v16h bH2 = *reinterpret_cast<const v16h*>(&whhP[((n * 4 + 2) * 32 + lane) * 8]);
          v16h bH3 = *reinterpret_cast<const v16h*>(&whhP[((n * 4 + 3) * 32 + lane) * 8]);
          float bx = bihN[cn], bh = bhhN[cn];
          v8f cx; for (int j = 0; j < 8; ++j) cx[j] = bx;
          v8f ch; for (int j = 0; j < 8; ++j) ch[j] = bh;
          cx = __builtin_amdgcn_wmma_f32_16x16x32_f16(false, aL,  false, bI,  (short)0, cx, false, false);
          ch = __builtin_amdgcn_wmma_f32_16x16x32_f16(false, aH0, false, bH0, (short)0, ch, false, false);
          ch = __builtin_amdgcn_wmma_f32_16x16x32_f16(false, aH1, false, bH1, (short)0, ch, false, false);
          ch = __builtin_amdgcn_wmma_f32_16x16x32_f16(false, aH2, false, bH2, (short)0, ch, false, false);
          ch = __builtin_amdgcn_wmma_f32_16x16x32_f16(false, aH3, false, bH3, (short)0, ch, false, false);
          unsigned* dx = reinterpret_cast<unsigned*>(&xnT[cn * 16 + mb]);
          dx[0] = pack2h(cx[0], cx[1]);
          dx[1] = pack2h(cx[2], cx[3]);
          dx[2] = pack2h(cx[4], cx[5]);
          dx[3] = pack2h(cx[6], cx[7]);
          unsigned* dh = reinterpret_cast<unsigned*>(&hnT[cn * 16 + mb]);
          dh[0] = pack2h(ch[0], ch[1]);
          dh[1] = pack2h(ch[2], ch[3]);
          dh[2] = pack2h(ch[4], ch[5]);
          dh[3] = pack2h(ch[6], ch[7]);
        }
      }
    }
    __syncthreads();

    // ---- Phase D: GRU elementwise update, h re-packed into WMMA A layout in place ----
    #pragma unroll
    for (int i = 0; i < 4; ++i) {
      const int dw = dwD[i], jb = jbD[i];
      float h0, h1; unpack2h(hA[dw], h0, h1);
      float r0 = (float)rzT[jb];
      float r1 = (float)rzT[jb + 16];
      float z0 = (float)rzT[2048 + jb];
      float z1 = (float)rzT[2048 + jb + 16];
      float n0 = tanh_fast((float)xnT[jb]      + r0 * (float)hnT[jb]);
      float n1 = tanh_fast((float)xnT[jb + 16] + r1 * (float)hnT[jb + 16]);
      float g0 = (1.f - z0) * n0 + z0 * h0;
      float g1 = (1.f - z1) * n1 + z1 * h1;
      hA[dw] = pack2h(g0, g1);
    }
    __syncthreads();

    // ---- Phase E: theta head as WMMA (wave 0 only; N padded 5->16) ----
    if (wave == 0) {
      v16h a0 = *reinterpret_cast<const v16h*>(&hA[(0 * 32 + lane) * 8]);
      v16h a1 = *reinterpret_cast<const v16h*>(&hA[(1 * 32 + lane) * 8]);
      v16h a2 = *reinterpret_cast<const v16h*>(&hA[(2 * 32 + lane) * 8]);
      v16h a3 = *reinterpret_cast<const v16h*>(&hA[(3 * 32 + lane) * 8]);
      v16h b0 = *reinterpret_cast<const v16h*>(&wheadP[(0 * 32 + lane) * 8]);
      v16h b1 = *reinterpret_cast<const v16h*>(&wheadP[(1 * 32 + lane) * 8]);
      v16h b2 = *reinterpret_cast<const v16h*>(&wheadP[(2 * 32 + lane) * 8]);
      v16h b3 = *reinterpret_cast<const v16h*>(&wheadP[(3 * 32 + lane) * 8]);
      int o = lane & 15;
      float bias = (o < 5) ? bheadS[o] : 0.f;
      v8f c; for (int j = 0; j < 8; ++j) c[j] = bias;
      c = __builtin_amdgcn_wmma_f32_16x16x32_f16(false, a0, false, b0, (short)0, c, false, false);
      c = __builtin_amdgcn_wmma_f32_16x16x32_f16(false, a1, false, b1, (short)0, c, false, false);
      c = __builtin_amdgcn_wmma_f32_16x16x32_f16(false, a2, false, b2, (short)0, c, false, false);
      c = __builtin_amdgcn_wmma_f32_16x16x32_f16(false, a3, false, b3, (short)0, c, false, false);
      if (o < 5) {
        #pragma unroll
        for (int j = 0; j < 8; ++j) {
          float th = 0.001f + 1.999f * sigmoidf_(c[j]);
          int m = j + mb;
          thetaS[m * 5 + o] = th;
          out_theta[((size_t)(bg * 16 + m) * KSTEPS + k) * 5 + o] = th;
        }
      }
    }
    __syncthreads();

    // ---- Phase F: input jump + RK4(5-state), write y_hat ----
    if (t < 16) {
      int m = t;
      float y[5], th[5], d1[5], d2[5], d3[5], d4[5], yt[5];
      #pragma unroll
      for (int s = 0; s < 5; ++s) { y[s] = yfull[m * 5 + s]; th[s] = thetaS[m * 5 + s]; }
      float u0 = uPtrF[0], u1 = uPtrF[1], u2v = uPtrF[2];
      y[0] += u0 * u2yS[0] + u1 * u2yS[3] + u2v * u2yS[6];
      y[2] += u0 * u2yS[1] + u1 * u2yS[4] + u2v * u2yS[7];
      y[4] += u0 * u2yS[2] + u1 * u2yS[5] + u2v * u2yS[8];
      float h = dtPtrF[0];                 // N_SUB = 1
      rhs5(y, th, d1);
      #pragma unroll
      for (int s = 0; s < 5; ++s) yt[s] = y[s] + 0.5f * h * d1[s];
      rhs5(yt, th, d2);
      #pragma unroll
      for (int s = 0; s < 5; ++s) yt[s] = y[s] + 0.5f * h * d2[s];
      rhs5(yt, th, d3);
      #pragma unroll
      for (int s = 0; s < 5; ++s) yt[s] = y[s] + h * d3[s];
      rhs5(yt, th, d4);
      #pragma unroll
      for (int s = 0; s < 5; ++s)
        y[s] = fmaxf(y[s] + (h / 6.f) * (d1[s] + 2.f * d2[s] + 2.f * d3[s] + d4[s]), 0.f);
      #pragma unroll
      for (int s = 0; s < 5; ++s) yfull[m * 5 + s] = y[s];
      yoPtrF[0] = y[0]; yoPtrF[1] = y[2]; yoPtrF[2] = y[4];
      uPtrF += 3; dtPtrF += 1; yoPtrF += 3;
    }
    __syncthreads();

    mod50 = (mod50 == 49) ? 0 : (mod50 + 1);
  }
}

extern "C" void kernel_launch(void* const* d_in, const int* in_sizes, int n_in,
                              void* d_out, int out_size, void* d_ws, size_t ws_size,
                              hipStream_t stream) {
  (void)in_sizes; (void)n_in; (void)d_ws; (void)ws_size; (void)out_size;
  const float* y0     = (const float*)d_in[0];
  const float* u_seq  = (const float*)d_in[1];
  const float* dt_seq = (const float*)d_in[2];
  const float* y_seq  = (const float*)d_in[3];
  const float* W_lift = (const float*)d_in[4];
  const float* b_lift = (const float*)d_in[5];
  const float* W_ih   = (const float*)d_in[6];
  const float* W_hh   = (const float*)d_in[7];
  const float* b_ih   = (const float*)d_in[8];
  const float* b_hh   = (const float*)d_in[9];
  const float* W_head = (const float*)d_in[10];
  const float* b_head = (const float*)d_in[11];
  const float* u2y    = (const float*)d_in[12];
  float* out        = (float*)d_out;
  float* out_yhat   = out;                                  // (512,512,3)
  float* out_theta  = out + (size_t)512 * 512 * 3;          // (512,512,5)
  kinetics_rnn_kernel<<<dim3(BATCH / 16), dim3(256), 0, stream>>>(
      y0, u_seq, dt_seq, y_seq, W_lift, b_lift, W_ih, W_hh, b_ih, b_hh,
      W_head, b_head, u2y, out_yhat, out_theta);
}
